// NarrativeClassificationLoss_80676665688521
// MI455X (gfx1250) — compile-verified, matrix operand order checked by reading, not run on hardware
//
#include <hip/hip_runtime.h>
#include <hip/hip_bf16.h>
#include <math.h>

// Problem constants (match reference)
#define BATCH   16384
#define NNARR   128
#define NSUB    1024
#define KGRP    8                    // NSUB / NNARR
#define COLS_TOTAL (NNARR + NSUB)    // 1152

// Workspace layout (in floats)
#define WS_COLSUM 0                          // [0,1152): per-class positive counts
#define WS_NPW    1152                       // [1152,1280): narrative pos_weight
#define WS_SPW    1280                       // [1280,2304): subnarrative pos_weight
#define WS_PART   2304                       // per-block partials, 5 per block
#define NBLOCKS   2048                       // main-pass grid
#define WS_FLOATS (WS_PART + 5 * NBLOCKS)    // 12544 floats ~ 50 KB

typedef _Float16 v16h __attribute__((ext_vector_type(16)));
typedef float    v8f  __attribute__((ext_vector_type(8)));

// One exp, one log1p -> softplus(x), softplus(-x), sigmoid(x).
//   e = exp(-|x|) in (0,1];  l = log1p(e)
//   softplus(x)  = max(x,0)  + l
//   softplus(-x) = max(-x,0) + l
//   sigmoid(x)   = x>=0 ? 1/(1+e) : e/(1+e)
__device__ __forceinline__ void sig_sp(float x, float& sp_px, float& sp_nx, float& p) {
    float e = __expf(-fabsf(x));
    float l = log1pf(e);
    sp_px = fmaxf(x, 0.0f) + l;
    sp_nx = fmaxf(-x, 0.0f) + l;
    float r = 1.0f / (1.0f + e);
    p = (x >= 0.0f) ? r : (e * r);
}

// ---------------------------------------------------------------------------
// 0) zero the workspace regions we accumulate into
// ---------------------------------------------------------------------------
__global__ void zero_ws_kernel(float* __restrict__ ws, int n) {
    int i = blockIdx.x * blockDim.x + threadIdx.x;
    if (i < n) ws[i] = 0.0f;
}

// ---------------------------------------------------------------------------
// 1) Column sums of an int32 0/1 label matrix via WMMA:
//    ones(16x32,f16) x labels_chunk(32x16,f16) accumulated into f32 C.
//    With A == all-ones, every row of D equals the column sums, so the
//    result only depends on B's column mapping N = lane&15 (ISA-guaranteed).
//    One wave per block; grid = (cols/16 tiles, row chunks).
// ---------------------------------------------------------------------------
__global__ __launch_bounds__(32)
void colsum_wmma_kernel(const int* __restrict__ labels, int cols,
                        float* __restrict__ out, int rows_per_block) {
    const int lane  = threadIdx.x;                // 0..31, wave32
    const int col   = blockIdx.x * 16 + (lane & 15);
    const int khalf = (lane >> 4) * 16;           // lanes 0-15: K 0..15, lanes 16-31: K 16..31
    const int row0  = blockIdx.y * rows_per_block;

    v16h a;
#pragma unroll
    for (int i = 0; i < 16; ++i) a[i] = (_Float16)1.0f;

    v8f c = {};
    const int iters = rows_per_block / 32;
    for (int it = 0; it < iters; ++it) {
        const int rbase = row0 + it * 32 + khalf;
        const int* p = labels + (size_t)rbase * cols + col;
        v16h b;
#pragma unroll
        for (int j = 0; j < 16; ++j)
            b[j] = (_Float16)(float)p[(size_t)j * cols];
        // D = A(ones) x B + C  -> every row of D = column sums of B chunk
        c = __builtin_amdgcn_wmma_f32_16x16x32_f16(
                /*neg_a=*/false, a, /*neg_b=*/false, b,
                /*c_mod=*/(short)0, c, /*reuse_a=*/false, /*reuse_b=*/false);
    }
    // integer-valued f32 sums < 2^24: atomicAdd is exact & order-independent
    if (lane < 16) atomicAdd(&out[col], c[0]);
}

// ---------------------------------------------------------------------------
// 2) pos_weight = clip((B - pos) / (pos + 1e-6), 1, 50); npw/spw contiguous
// ---------------------------------------------------------------------------
__global__ void compute_pw_kernel(float* __restrict__ ws) {
    int i = blockIdx.x * blockDim.x + threadIdx.x;
    if (i < COLS_TOTAL) {
        float pos = ws[WS_COLSUM + i];
        float neg = (float)BATCH - pos;
        float pw  = neg / (pos + 1e-6f);
        ws[WS_NPW + i] = fminf(fmaxf(pw, 1.0f), 50.0f);
    }
}

// ---------------------------------------------------------------------------
// 3) Fused main pass: thread t owns narrative class t and sub group
//    [t*8, t*8+8). Grid-stride over rows; per-block partials of
//    {bce_n, focal_n, masked_group_bce, focal_s, hierarchy}.
// ---------------------------------------------------------------------------
__global__ __launch_bounds__(NNARR)
void main_loss_kernel(const float* __restrict__ nlog, const float* __restrict__ slog,
                      const int* __restrict__ nlab, const int* __restrict__ slab,
                      const float* __restrict__ ws, float* __restrict__ part) {
    const int t = threadIdx.x;  // 0..127

    const float npw_t = ws[WS_NPW + t];
    float spw_t[KGRP];
#pragma unroll
    for (int j = 0; j < KGRP; ++j) spw_t[j] = ws[WS_SPW + t * KGRP + j];

    float accA = 0.f, accFn = 0.f, accS = 0.f, accFs = 0.f, accH = 0.f;

    for (int row = blockIdx.x; row < BATCH; row += gridDim.x) {
        // ---- narrative element (col t) ----
        float x = nlog[(size_t)row * NNARR + t];
        float y = (float)nlab[(size_t)row * NNARR + t];
        float sp_px, sp_nx, p;
        sig_sp(x, sp_px, sp_nx, p);
        accA += npw_t * y * sp_nx + (1.0f - y) * sp_px;
        float om = 1.0f - p;
        accFn += om * om * y * (-sp_nx);          // log_sigmoid(x) = -softplus(-x)

        // ---- sub group of 8 (128-bit loads) ----
        const float4* sx4 = (const float4*)(slog + (size_t)row * NSUB + t * KGRP);
        const int4*   sy4 = (const int4*)(slab + (size_t)row * NSUB + t * KGRP);
        float4 xa = sx4[0], xb = sx4[1];
        int4   ya = sy4[0], yb = sy4[1];
        float xs[KGRP] = {xa.x, xa.y, xa.z, xa.w, xb.x, xb.y, xb.z, xb.w};
        float ys[KGRP] = {(float)ya.x, (float)ya.y, (float)ya.z, (float)ya.w,
                          (float)yb.x, (float)yb.y, (float)yb.z, (float)yb.w};

        float gsum = 0.f, gmax = 0.f;             // probs > 0, so 0 is a safe lower bound
#pragma unroll
        for (int j = 0; j < KGRP; ++j) {
            float spp, spn, ps;
            sig_sp(xs[j], spp, spn, ps);
            gsum += spw_t[j] * ys[j] * spn + (1.0f - ys[j]) * spp;
            float omp = 1.0f - ps;
            accFs += omp * omp * ys[j] * (-spn);
            gmax = fmaxf(gmax, ps);
        }
        accS += y * (gsum * (1.0f / (float)KGRP));
        accH += y * fmaxf(gmax - p, 0.0f);
    }

    __shared__ float red[NNARR];
    float vals[5] = {accA, accFn, accS, accFs, accH};
    float sums[5];
#pragma unroll
    for (int k = 0; k < 5; ++k) {
        red[t] = vals[k];
        __syncthreads();
        for (int s = NNARR / 2; s > 0; s >>= 1) {
            if (t < s) red[t] += red[t + s];
            __syncthreads();
        }
        sums[k] = red[0];
        __syncthreads();
    }
    if (t == 0) {
        float* pp = part + (size_t)blockIdx.x * 5;
#pragma unroll
        for (int k = 0; k < 5; ++k) pp[k] = sums[k];
    }
}

// ---------------------------------------------------------------------------
// 4) Deterministic finalize: reduce per-block partials + valid_count, combine
// ---------------------------------------------------------------------------
__global__ __launch_bounds__(256)
void finalize_kernel(const float* __restrict__ ws, float* __restrict__ out) {
    __shared__ float red[256];
    const int t = threadIdx.x;

    float a[5] = {0.f, 0.f, 0.f, 0.f, 0.f};
    for (int b = t; b < NBLOCKS; b += 256) {
        const float* pp = ws + WS_PART + (size_t)b * 5;
#pragma unroll
        for (int k = 0; k < 5; ++k) a[k] += pp[k];
    }
    float s[5];
#pragma unroll
    for (int k = 0; k < 5; ++k) {
        red[t] = a[k];
        __syncthreads();
        for (int st = 128; st > 0; st >>= 1) {
            if (t < st) red[t] += red[t + st];
            __syncthreads();
        }
        s[k] = red[0];
        __syncthreads();
    }
    // valid_count = total positive narrative labels = sum of narrative colsums
    red[t] = (t < NNARR) ? ws[WS_COLSUM + t] : 0.0f;
    __syncthreads();
    for (int st = 128; st > 0; st >>= 1) {
        if (t < st) red[t] += red[t + st];
        __syncthreads();
    }
    float vc = red[0];

    if (t == 0) {
        const float invNE = 1.0f / ((float)BATCH * (float)NNARR);
        const float invSE = 1.0f / ((float)BATCH * (float)NSUB);
        float narrative_loss = s[0] * invNE;
        float nf             = s[1] * invNE;
        float sub_loss       = (vc > 0.0f) ? (s[2] / fmaxf(vc, 1.0f)) : 0.0f;
        float sf             = s[3] * invSE;
        float hier           = s[4] / (float)BATCH;
        // NW*(narr - 0.1*nf) + SW*(sub - 0.1*sf) + HW*hier ; NW=SW=1, HW=0.5
        out[0] = (narrative_loss - 0.1f * nf)
               + (sub_loss       - 0.1f * sf)
               + 0.5f * hier;
    }
}

// ---------------------------------------------------------------------------
extern "C" void kernel_launch(void* const* d_in, const int* in_sizes, int n_in,
                              void* d_out, int out_size, void* d_ws, size_t ws_size,
                              hipStream_t stream) {
    const float* nlog = (const float*)d_in[0];   // [B, 128]
    const float* slog = (const float*)d_in[1];   // [B, 1024]
    const int*   nlab = (const int*)d_in[2];     // [B, 128]
    const int*   slab = (const int*)d_in[3];     // [B, 1024]
    float* ws  = (float*)d_ws;
    float* out = (float*)d_out;

    (void)in_sizes; (void)n_in; (void)out_size; (void)ws_size;

    // 0) zero accumulation regions
    zero_ws_kernel<<<(WS_FLOATS + 255) / 256, 256, 0, stream>>>(ws, WS_FLOATS);

    // 1) WMMA column sums (rows split 64 ways, 256 rows/block -> 8 WMMAs each)
    colsum_wmma_kernel<<<dim3(NNARR / 16, 64), 32, 0, stream>>>(
        nlab, NNARR, ws + WS_COLSUM, BATCH / 64);
    colsum_wmma_kernel<<<dim3(NSUB / 16, 64), 32, 0, stream>>>(
        slab, NSUB, ws + WS_COLSUM + NNARR, BATCH / 64);

    // 2) pos_weights
    compute_pw_kernel<<<(COLS_TOTAL + 255) / 256, 256, 0, stream>>>(ws);

    // 3) fused BCE / focal / group / hierarchy pass
    main_loss_kernel<<<NBLOCKS, NNARR, 0, stream>>>(
        nlog, slog, nlab, slab, ws, ws + WS_PART);

    // 4) finalize scalar
    finalize_kernel<<<1, 256, 0, stream>>>(ws, out);
}
